// ECN3_85237920956778
// MI455X (gfx1250) — compile-verified, matrix operand order checked by reading, not run on hardware
//
#include <hip/hip_runtime.h>
#include <hip/hip_bf16.h>
#include <stdint.h>

// ---------------------------------------------------------------------------
// Types for CDNA5 WMMA (wave32): v_wmma_f32_16x16x32_bf16
// ---------------------------------------------------------------------------
typedef __bf16 bf16_t;
typedef __attribute__((ext_vector_type(16))) __bf16 v16bf;
typedef __attribute__((ext_vector_type(8)))  float  v8f;
typedef __attribute__((ext_vector_type(4)))  unsigned int v4u;

union FragAB { v16bf v; v4u q[2]; };

static constexpr int Gg = 64;
static constexpr int Nn = 256;

// ---------------------------------------------------------------------------
// Zero helper (ws is poisoned by harness; stats must start at 0 each call)
// ---------------------------------------------------------------------------
__global__ void zero_f32(float* p, int n) {
  int i = blockIdx.x * blockDim.x + threadIdx.x;
  if (i < n) p[i] = 0.f;
}

// ---------------------------------------------------------------------------
// kNN: one block (256 threads) per graph. Features in LDS, register top-k.
// ---------------------------------------------------------------------------
template <int KK, int F>
__global__ __launch_bounds__(256) void knn_kernel(const float* __restrict__ X,
                                                  int* __restrict__ out) {
  extern __shared__ float xs[];
  const int g = blockIdx.x;
  const float* Xg = X + (size_t)g * Nn * F;
  for (int t = threadIdx.x; t < Nn * F; t += blockDim.x) xs[t] = Xg[t];
  __syncthreads();

  const int i = threadIdx.x;  // blockDim == Nn == 256
  float bd[KK];
  int   bi[KK];
#pragma unroll
  for (int t = 0; t < KK; ++t) { bd[t] = 3.4e38f; bi[t] = 0; }

  for (int j = 0; j < Nn; ++j) {
    float d2 = 0.f;
    if constexpr ((F & 3) == 0) {
      const float4* xi = (const float4*)(xs + i * F);
      const float4* xj = (const float4*)(xs + j * F);
      for (int c = 0; c < F / 4; ++c) {
        float4 a = xi[c], b = xj[c];
        float dx = a.x - b.x, dy = a.y - b.y, dz = a.z - b.z, dw = a.w - b.w;
        d2 += dx * dx + dy * dy + dz * dz + dw * dw;
      }
    } else {
      for (int c = 0; c < F; ++c) {
        float df = xs[i * F + c] - xs[j * F + c];
        d2 += df * df;
      }
    }
    if (j == i) d2 = 1e10f;  // reference adds 1e10 on the diagonal
    if (d2 < bd[KK - 1]) {
      bd[KK - 1] = d2; bi[KK - 1] = j;
#pragma unroll
      for (int t = KK - 1; t > 0; --t) {
        if (bd[t] < bd[t - 1]) {
          float td = bd[t]; bd[t] = bd[t - 1]; bd[t - 1] = td;
          int   ti = bi[t]; bi[t] = bi[t - 1]; bi[t - 1] = ti;
        }
      }
    }
  }
  int* og = out + ((size_t)g * Nn + i) * KK;
#pragma unroll
  for (int t = 0; t < KK; ++t) og[t] = bi[t];
}

// ---------------------------------------------------------------------------
// Pack W (K2 x F, row-major f32) into bf16 WMMA B-fragment order, K padded to
// K2P (multiple of 32). Fragment (kt,nt), lane L, elem t:
//   col N = nt*16 + (L&15), kbase = (L>>4)*8,
//   K = kt*32 + (t<8 ? kbase+t : kbase+16+(t-8))
// Stored contiguously: Wp[((kt*NT+nt)*32 + L)*16 + t]  -> two b128 loads/lane.
// ---------------------------------------------------------------------------
__global__ void pack_w(const float* __restrict__ W, bf16_t* __restrict__ Wp,
                       int K2, int K2P, int F) {
  const int NT = F / 16;
  const int total = (K2P / 32) * NT * 32 * 16;
  int id = blockIdx.x * blockDim.x + threadIdx.x;
  if (id >= total) return;
  int t    = id & 15;
  int lane = (id >> 4) & 31;
  int fr   = id >> 9;
  int nt   = fr % NT;
  int kt   = fr / NT;
  int kb   = (lane >> 4) * 8;
  int kin  = (t < 8) ? (kb + t) : (kb + 16 + (t - 8));
  int k    = kt * 32 + kin;
  int n    = nt * 16 + (lane & 15);
  float v  = (k < K2) ? W[(size_t)k * F + n] : 0.f;
  Wp[id] = (bf16_t)v;
}

// ---------------------------------------------------------------------------
// EdgeConv main: one block = one (graph, 16-point tile). 8 waves, wave32.
// Builds edge matrix M = [x_i, x_j - x_i] (16*KN x K2P) in LDS as bf16,
// then 2x2 register-tiled WMMA bf16 GEMM against packed W:
//   work item = (M-tile pair, N-tile pair) -> 2 A-frags + 2 B-frags per K-step
//   feed 4 independent WMMAs (halves LDS and L2 bytes per WMMA, hides the
//   WMMA->WMMA RAW hazard). ReLU then:
//   - sum over k neighbors per (point, channel) -> LDS ds_add_f32 -> global
//   - per-channel sum / sum-of-squares for BN    -> LDS -> one global atomic
// ---------------------------------------------------------------------------
template <int FIN, int KN, int K2P, int FOUT>
__global__ __launch_bounds__(256) void ec_main(
    const float* __restrict__ X, const int* __restrict__ nbr,
    const bf16_t* __restrict__ Wp, const float* __restrict__ bias,
    float* __restrict__ sumk_out, float* __restrict__ gs1,
    float* __restrict__ gs2) {
  constexpr int EDGES = 16 * KN;
  constexpr int KT = K2P / 32;
  constexpr int NT = FOUT / 16;
  constexpr int MP = KN / 2;  // M-tile pairs (KN is 8/10/12: always even)
  constexpr int NP = NT / 2;  // N-tile pairs

  extern __shared__ char smem[];
  bf16_t* mt   = (bf16_t*)smem;                                        // EDGES*K2P
  float*  sumk = (float*)(smem + (size_t)EDGES * K2P * sizeof(bf16_t)); // 16*FOUT
  float*  bs1  = sumk + 16 * FOUT;                                      // FOUT
  float*  bs2  = bs1 + FOUT;                                            // FOUT

  const int tid   = threadIdx.x;
  const int g     = blockIdx.x / (Nn / 16);
  const int pbase = (blockIdx.x % (Nn / 16)) * 16;

  for (int i = tid; i < 16 * FOUT + 2 * FOUT; i += blockDim.x) sumk[i] = 0.f;

  const float* Xg   = X + (size_t)g * Nn * FIN;
  const int*   idxg = nbr + ((size_t)g * Nn + pbase) * KN;

  // Build bf16 edge-feature tile in LDS.
  for (int t = tid; t < EDGES * K2P; t += blockDim.x) {
    int e  = t / K2P;
    int c  = t - e * K2P;
    int pt = e / KN;
    int kk = e - pt * KN;
    int i  = pbase + pt;
    float v = 0.f;
    if (c < FIN) {
      v = Xg[(size_t)i * FIN + c];
    } else if (c < 2 * FIN) {
      int j  = idxg[pt * KN + kk];
      int cc = c - FIN;
      v = Xg[(size_t)j * FIN + cc] - Xg[(size_t)i * FIN + cc];
    }
    mt[t] = (bf16_t)v;
  }
  __syncthreads();

  const int wave = tid >> 5;
  const int lane = tid & 31;
  const int row  = lane & 15;        // A row / D column index
  const int kb   = (lane >> 4) * 8;  // A/B K base for this half-wave

  for (int w = wave; w < MP * NP; w += 8) {
    const int m0 = (w % MP) * 2;
    const int m1 = m0 + 1;
    const int n0 = (w / MP) * 2;
    const int n1 = n0 + 1;

    const bf16_t* arow0 = mt + (size_t)(m0 * 16 + row) * K2P;
    const bf16_t* arow1 = mt + (size_t)(m1 * 16 + row) * K2P;

    float bv0 = bias[n0 * 16 + row];
    float bv1 = bias[n1 * 16 + row];
    v8f acc00 = {bv0, bv0, bv0, bv0, bv0, bv0, bv0, bv0};
    v8f acc01 = {bv1, bv1, bv1, bv1, bv1, bv1, bv1, bv1};
    v8f acc10 = acc00;
    v8f acc11 = acc01;

    for (int kt = 0; kt < KT; ++kt) {
      FragAB a0, a1, b0, b1;
      const v4u* ap0 = (const v4u*)(arow0 + kt * 32 + kb);
      const v4u* ap1 = (const v4u*)(arow1 + kt * 32 + kb);
      a0.q[0] = ap0[0];  // K = kb+0..kb+7   (elems 0..7)
      a0.q[1] = ap0[2];  // K = kb+16..kb+23 (elems 8..15)
      a1.q[0] = ap1[0];
      a1.q[1] = ap1[2];
      const v4u* bp0 =
          (const v4u*)(Wp + ((((size_t)kt * NT + n0) * 32 + lane) * 16));
      const v4u* bp1 =
          (const v4u*)(Wp + ((((size_t)kt * NT + n1) * 32 + lane) * 16));
      b0.q[0] = bp0[0];
      b0.q[1] = bp0[1];
      b1.q[0] = bp1[0];
      b1.q[1] = bp1[1];
      acc00 = __builtin_amdgcn_wmma_f32_16x16x32_bf16(
          false, a0.v, false, b0.v, (short)0, acc00, false, false);
      acc01 = __builtin_amdgcn_wmma_f32_16x16x32_bf16(
          false, a0.v, false, b1.v, (short)0, acc01, false, false);
      acc10 = __builtin_amdgcn_wmma_f32_16x16x32_bf16(
          false, a1.v, false, b0.v, (short)0, acc10, false, false);
      acc11 = __builtin_amdgcn_wmma_f32_16x16x32_bf16(
          false, a1.v, false, b1.v, (short)0, acc11, false, false);
    }

    // ReLU + reductions. Lane holds one column (channel) across 8 rows.
    const int Mofs = (lane >> 4) << 3;
    v8f accs[4] = {acc00, acc01, acc10, acc11};
    const int mids[4] = {m0, m0, m1, m1};
    const int nids[4] = {n0, n1, n0, n1};
#pragma unroll
    for (int u = 0; u < 4; ++u) {
      float s1 = 0.f, s2 = 0.f;
      const int mtile = mids[u];
      const int col   = nids[u] * 16 + row;
#pragma unroll
      for (int r = 0; r < 8; ++r) {
        float h = accs[u][r] > 0.f ? accs[u][r] : 0.f;
        s1 += h;
        s2 += h * h;
        int eloc = mtile * 16 + (r + Mofs);
        int pt   = eloc / KN;
        atomicAdd(&sumk[pt * FOUT + col], h);
      }
      atomicAdd(&bs1[col], s1);
      atomicAdd(&bs2[col], s2);
    }
  }
  __syncthreads();

  float* outg = sumk_out + ((size_t)g * Nn + pbase) * FOUT;
  for (int t = tid; t < 16 * FOUT; t += blockDim.x) outg[t] = sumk[t];
  for (int f = tid; f < FOUT; f += blockDim.x) {
    atomicAdd(&gs1[f], bs1[f]);
    atomicAdd(&gs2[f], bs2[f]);
  }
}

// ---------------------------------------------------------------------------
// BN-finish + mean over k (affine BN commutes with mean over k):
//   x <- scale * (sumk / KN) + shift
// ---------------------------------------------------------------------------
__global__ void ec_finish(float* __restrict__ x, const float* __restrict__ gs1,
                          const float* __restrict__ gs2,
                          const float* __restrict__ gamma,
                          const float* __restrict__ beta, int total, int F,
                          float invcount, float invK) {
  int i = blockIdx.x * blockDim.x + threadIdx.x;
  if (i >= total) return;
  int f = i % F;
  float mu  = gs1[f] * invcount;
  float var = gs2[f] * invcount - mu * mu;
  float sc  = gamma[f] * rsqrtf(var + 1e-5f);
  float sh  = beta[f] - mu * sc;
  x[i] = sc * (x[i] * invK) + sh;
}

// ---------------------------------------------------------------------------
// Global mean pool over points: p[g,f] = mean_n x[g,n,f]
// ---------------------------------------------------------------------------
__global__ void pool_mean(const float* __restrict__ x, float* __restrict__ p,
                          int F) {
  int i = blockIdx.x * blockDim.x + threadIdx.x;  // over G*F
  if (i >= Gg * F) return;
  int g = i / F;
  int f = i - g * F;
  float s = 0.f;
  const float* xg = x + (size_t)g * Nn * F + f;
  for (int n = 0; n < Nn; ++n) s += xg[(size_t)n * F];
  p[i] = s * (1.f / Nn);
}

// ---------------------------------------------------------------------------
// FC1: r1[g,f] = relu(p[g,:] @ Wc1[:,f] + bc1[f]); channel stats via atomics.
// Block = (g, 256 channels); p row staged in LDS.
// ---------------------------------------------------------------------------
__global__ __launch_bounds__(256) void fc1_kernel(
    const float* __restrict__ p, const float* __restrict__ Wc1,
    const float* __restrict__ bc1, float* __restrict__ r1,
    float* __restrict__ s1, float* __restrict__ s2, int F) {
  __shared__ float pl[512];
  int g = blockIdx.x >> 1;
  int f = ((blockIdx.x & 1) << 8) + threadIdx.x;
  for (int t = threadIdx.x; t < F; t += 256) pl[t] = p[(size_t)g * F + t];
  __syncthreads();
  float acc = bc1[f];
  for (int c = 0; c < F; ++c) acc += pl[c] * Wc1[(size_t)c * F + f];
  float h = fmaxf(acc, 0.f);
  r1[(size_t)g * F + f] = h;
  atomicAdd(&s1[f], h);
  atomicAdd(&s2[f], h * h);
}

// ---------------------------------------------------------------------------
// FC2 + both final BNs + sigmoid. Single block (G=64 fits in one block).
// ---------------------------------------------------------------------------
__global__ __launch_bounds__(256) void fc2_kernel(
    const float* __restrict__ r1, const float* __restrict__ s1,
    const float* __restrict__ s2, const float* __restrict__ gc1,
    const float* __restrict__ bec1, const float* __restrict__ Wc2,
    const float* __restrict__ bc2, const float* __restrict__ gc2,
    const float* __restrict__ bec2, float* __restrict__ out, int F) {
  __shared__ float z[Gg];
  __shared__ float red[2];
  int tid = threadIdx.x;
  if (tid < Gg) {
    float acc = bc2[0];
    for (int c = 0; c < F; ++c) {
      float mu  = s1[c] * (1.f / Gg);
      float var = s2[c] * (1.f / Gg) - mu * mu;
      float sc  = gc1[c] * rsqrtf(var + 1e-5f);
      float hn  = sc * (r1[(size_t)tid * F + c] - mu) + bec1[c];
      acc += hn * Wc2[c];
    }
    z[tid] = fmaxf(acc, 0.f);
  }
  __syncthreads();
  if (tid == 0) {
    float m = 0.f, v = 0.f;
    for (int g = 0; g < Gg; ++g) m += z[g];
    m *= (1.f / Gg);
    for (int g = 0; g < Gg; ++g) { float d = z[g] - m; v += d * d; }
    v *= (1.f / Gg);
    red[0] = m;
    red[1] = rsqrtf(v + 1e-5f);
  }
  __syncthreads();
  if (tid < Gg) {
    float val = gc2[0] * (z[tid] - red[0]) * red[1] + bec2[0];
    out[tid] = 1.f / (1.f + expf(-val));
  }
}

// ---------------------------------------------------------------------------
// Host-side orchestration
// ---------------------------------------------------------------------------
extern "C" void kernel_launch(void* const* d_in, const int* in_sizes, int n_in,
                              void* d_out, int out_size, void* d_ws,
                              size_t ws_size, hipStream_t stream) {
  (void)in_sizes; (void)n_in; (void)out_size; (void)ws_size;

  const float* x    = (const float*)d_in[0];
  const float* pos  = (const float*)d_in[1];
  const float* W1   = (const float*)d_in[3];
  const float* b1   = (const float*)d_in[4];
  const float* g1   = (const float*)d_in[5];
  const float* be1  = (const float*)d_in[6];
  const float* W2   = (const float*)d_in[7];
  const float* b2   = (const float*)d_in[8];
  const float* g2   = (const float*)d_in[9];
  const float* be2  = (const float*)d_in[10];
  const float* W3   = (const float*)d_in[11];
  const float* b3   = (const float*)d_in[12];
  const float* g3   = (const float*)d_in[13];
  const float* be3  = (const float*)d_in[14];
  const float* Wc1  = (const float*)d_in[15];
  const float* bc1  = (const float*)d_in[16];
  const float* gc1  = (const float*)d_in[17];
  const float* bec1 = (const float*)d_in[18];
  const float* Wc2  = (const float*)d_in[19];
  const float* bc2  = (const float*)d_in[20];
  const float* gc2  = (const float*)d_in[21];
  const float* bec2 = (const float*)d_in[22];
  float* out = (float*)d_out;

  // Workspace carve-up (256B aligned).
  char* ws = (char*)d_ws;
  size_t off = 0;
  auto carve = [&](size_t bytes) {
    char* p = ws + off;
    off += (bytes + 255) & ~(size_t)255;
    return p;
  };
  float*  bufA  = (float*)carve((size_t)Gg * Nn * 512 * 4);
  float*  bufB  = (float*)carve((size_t)Gg * Nn * 512 * 4);
  int*    idxb  = (int*)carve((size_t)Gg * Nn * 12 * 4);
  bf16_t* Wp1   = (bf16_t*)carve((size_t)(96 / 32) * (128 / 16) * 512 * 2);
  bf16_t* Wp2   = (bf16_t*)carve((size_t)(256 / 32) * (256 / 16) * 512 * 2);
  bf16_t* Wp3   = (bf16_t*)carve((size_t)(512 / 32) * (512 / 16) * 512 * 2);
  float*  stats = (float*)carve(2 * 512 * 4);
  float*  pbuf  = (float*)carve((size_t)Gg * 512 * 4);
  float*  r1buf = (float*)carve((size_t)Gg * 512 * 4);

  // --- Pack weights to bf16 fragment order ---
  pack_w<<<(3 * 8 * 512 + 255) / 256, 256, 0, stream>>>(W1, Wp1, 70, 96, 128);
  pack_w<<<(8 * 16 * 512 + 255) / 256, 256, 0, stream>>>(W2, Wp2, 256, 256, 256);
  pack_w<<<(16 * 32 * 512 + 255) / 256, 256, 0, stream>>>(W3, Wp3, 512, 512, 512);

  const int ecGrid = Gg * (Nn / 16);

  // --- Stage 1: knn(pos,8) + EdgeConv(35 -> 128) ---
  knn_kernel<8, 3><<<Gg, 256, Nn * 3 * 4, stream>>>(pos, idxb);
  zero_f32<<<1, 256, 0, stream>>>(stats, 2 * 128);
  {
    size_t lds = (size_t)(16 * 8) * 96 * 2 + (16 + 2) * 128 * 4;
    ec_main<35, 8, 96, 128><<<ecGrid, 256, lds, stream>>>(
        x, idxb, Wp1, b1, bufA, stats, stats + 128);
  }
  ec_finish<<<(Gg * Nn * 128) / 256, 256, 0, stream>>>(
      bufA, stats, stats + 128, g1, be1, Gg * Nn * 128, 128,
      1.f / (Gg * Nn * 8), 1.f / 8.f);

  // --- Stage 2: knn(x1,10) + EdgeConv(128 -> 256) ---
  knn_kernel<10, 128><<<Gg, 256, Nn * 128 * 4, stream>>>(bufA, idxb);
  zero_f32<<<2, 256, 0, stream>>>(stats, 2 * 256);
  {
    size_t lds = (size_t)(16 * 10) * 256 * 2 + (16 + 2) * 256 * 4;
    ec_main<128, 10, 256, 256><<<ecGrid, 256, lds, stream>>>(
        bufA, idxb, Wp2, b2, bufB, stats, stats + 256);
  }
  ec_finish<<<(Gg * Nn * 256) / 256, 256, 0, stream>>>(
      bufB, stats, stats + 256, g2, be2, Gg * Nn * 256, 256,
      1.f / (Gg * Nn * 10), 1.f / 10.f);

  // --- Stage 3: knn(x2,12) + EdgeConv(256 -> 512) ---
  knn_kernel<12, 256><<<Gg, 256, Nn * 256 * 4, stream>>>(bufB, idxb);
  zero_f32<<<4, 256, 0, stream>>>(stats, 2 * 512);
  {
    size_t lds = (size_t)(16 * 12) * 512 * 2 + (16 + 2) * 512 * 4;
    ec_main<256, 12, 512, 512><<<ecGrid, 256, lds, stream>>>(
        bufB, idxb, Wp3, b3, bufA, stats, stats + 512);
  }
  ec_finish<<<(Gg * Nn * 512) / 256, 256, 0, stream>>>(
      bufA, stats, stats + 512, g3, be3, Gg * Nn * 512, 512,
      1.f / (Gg * Nn * 12), 1.f / 12.f);

  // --- Head: pool -> FC1(+BN) -> FC2(+BN) -> sigmoid ---
  pool_mean<<<(Gg * 512 + 255) / 256, 256, 0, stream>>>(bufA, pbuf, 512);
  zero_f32<<<4, 256, 0, stream>>>(stats, 2 * 512);
  fc1_kernel<<<Gg * 2, 256, 0, stream>>>(pbuf, Wc1, bc1, r1buf, stats,
                                         stats + 512, 512);
  fc2_kernel<<<1, 256, 0, stream>>>(r1buf, stats, stats + 512, gc1, bec1, Wc2,
                                    bc2, gc2, bec2, out, 512);
}
